// LI_FusionModule_ClassAware_20151986553441
// MI455X (gfx1250) — matrix operand order repaired and checked by reading, not compiled, not run on hardware
//
#include <hip/hip_runtime.h>
#include <hip/hip_bf16.h>
#include <cstdint>

// ---------------------------------------------------------------------------
// LI fusion (class-aware) for gfx1250:
//   - all three GEMMs on v_wmma_f32_16x16x32_f16 (f16 in, f32 acc)
//   - image GEMM B-tile staged once per block in LDS (kills 8x redundancy)
//   - voxel-feature tile fetched with global_load_async_to_lds_b128 and
//     overlapped with projection/box-test/bilinear stages (ASYNCcnt)
// ---------------------------------------------------------------------------

typedef _Float16 v16h __attribute__((ext_vector_type(16)));
typedef _Float16 v8h  __attribute__((ext_vector_type(8)));
typedef _Float16 h4   __attribute__((ext_vector_type(4)));
typedef float    v8f  __attribute__((ext_vector_type(8)));

constexpr int NVOX = 262144;
constexpr int BB   = 4;
constexpr int MBOX = 32;
constexpr int C3D  = 64;
constexpr int C2D  = 256;
constexpr int MIDC = 128;
constexpr int OUTC = 64;
constexpr int HH   = 96;
constexpr int WW   = 312;
constexpr int HW   = HH * WW;          // 29952 (multiple of 16)

constexpr int FSTR = 264;              // fused LDS row stride (halves), padded
constexpr int VSTR = 68;               // vf LDS row stride (floats), padded

__device__ __forceinline__ v8f wmma16(v16h a, v16h b, v8f c) {
    return __builtin_amdgcn_wmma_f32_16x16x32_f16(
        /*neg_a=*/false, a, /*neg_b=*/false, b,
        /*c_mod=*/(short)0, c, /*reuse_a=*/false, /*reuse_b=*/false);
}

// ===========================================================================
// Kernel 0: BN folding + weight conversion / transposition (tiny)
// ===========================================================================
__global__ void prep_kernel(
    const float* __restrict__ w_vt, const float* __restrict__ w_it,
    const float* __restrict__ w_f,
    const float* __restrict__ b_vt, const float* __restrict__ g_vt,
    const float* __restrict__ be_vt, const float* __restrict__ m_vt,
    const float* __restrict__ v_vt,
    const float* __restrict__ b_it, const float* __restrict__ g_it,
    const float* __restrict__ be_it, const float* __restrict__ m_it,
    const float* __restrict__ v_it,
    const float* __restrict__ b_f, const float* __restrict__ g_f,
    const float* __restrict__ be_f, const float* __restrict__ m_f,
    const float* __restrict__ v_f,
    _Float16* __restrict__ w_it_h, _Float16* __restrict__ wvtT,
    _Float16* __restrict__ wfT,
    float* __restrict__ svt, float* __restrict__ hvt,
    float* __restrict__ sit, float* __restrict__ hit,
    float* __restrict__ sf,  float* __restrict__ hf)
{
    const int tid = blockIdx.x * blockDim.x + threadIdx.x;
    const int stride = gridDim.x * blockDim.x;
    for (int i = tid; i < MIDC * C2D; i += stride)
        w_it_h[i] = (_Float16)w_it[i];
    for (int i = tid; i < C3D * MIDC; i += stride) {
        int k = i / MIDC, o = i % MIDC;
        wvtT[i] = (_Float16)w_vt[o * C3D + k];
    }
    for (int i = tid; i < 2 * MIDC * OUTC; i += stride) {
        int k = i / OUTC, o = i % OUTC;
        wfT[i] = (_Float16)w_f[o * (2 * MIDC) + k];
    }
    for (int i = tid; i < MIDC; i += stride) {
        float s = g_vt[i] * rsqrtf(v_vt[i] + 1e-5f);
        svt[i] = s; hvt[i] = be_vt[i] + (b_vt[i] - m_vt[i]) * s;
    }
    for (int i = tid; i < MIDC; i += stride) {
        float s = g_it[i] * rsqrtf(v_it[i] + 1e-5f);
        sit[i] = s; hit[i] = be_it[i] + (b_it[i] - m_it[i]) * s;
    }
    for (int i = tid; i < OUTC; i += stride) {
        float s = g_f[i] * rsqrtf(v_f[i] + 1e-5f);
        sf[i] = s; hf[i] = be_f[i] + (b_f[i] - m_f[i]) * s;
    }
}

// ===========================================================================
// Kernel 1: image 1x1-conv + BN + ReLU via WMMA, B tile staged in LDS.
//   Block: 256 threads = 8 waves. Each block: 16 pixels x 128 channels.
// ===========================================================================
__global__ __launch_bounds__(256) void img_xform_kernel(
    const float* __restrict__ img,        // [B][C2D][HW]
    const _Float16* __restrict__ w_it_h,  // [MIDC][C2D]
    const float* __restrict__ sit, const float* __restrict__ hit,
    _Float16* __restrict__ it)            // [B][HW][MIDC]
{
    // B tile: [256 channels][16 px] f16, rows padded to 24 halves (48 B)
    __shared__ __align__(16) _Float16 btile[C2D * 24];

    const int wave  = threadIdx.x >> 5;
    const int lane  = threadIdx.x & 31;
    const int tiles = HW / 16;                 // 1872
    const int b     = blockIdx.x / tiles;
    const int p0    = (blockIdx.x % tiles) * 16;
    const int m0    = wave * 16;
    const int hi    = lane >> 4;
    const int mrow  = lane & 15;

    // ---- cooperative fill: thread t converts channel row t ----------------
    {
        const int c = threadIdx.x;                  // 0..255
        const float* xp = img + ((size_t)b * C2D + c) * HW + p0;
        float4 x0 = *(const float4*)(xp + 0);
        float4 x1 = *(const float4*)(xp + 4);
        float4 x2 = *(const float4*)(xp + 8);
        float4 x3 = *(const float4*)(xp + 12);
        v8h lo, hx;
        lo[0]=(_Float16)x0.x; lo[1]=(_Float16)x0.y; lo[2]=(_Float16)x0.z; lo[3]=(_Float16)x0.w;
        lo[4]=(_Float16)x1.x; lo[5]=(_Float16)x1.y; lo[6]=(_Float16)x1.z; lo[7]=(_Float16)x1.w;
        hx[0]=(_Float16)x2.x; hx[1]=(_Float16)x2.y; hx[2]=(_Float16)x2.z; hx[3]=(_Float16)x2.w;
        hx[4]=(_Float16)x3.x; hx[5]=(_Float16)x3.y; hx[6]=(_Float16)x3.z; hx[7]=(_Float16)x3.w;
        *(v8h*)&btile[c * 24]     = lo;
        *(v8h*)&btile[c * 24 + 8] = hx;
    }
    __syncthreads();

    v8f acc = {};
    const _Float16* wrow = w_it_h + (size_t)(m0 + mrow) * C2D;

    for (int c0 = 0; c0 < C2D; c0 += 32) {
        // A fragment (weights): K runs [c0+8h..+7] and [c0+16+8h..+7]
        const int kb = c0 + hi * 8;
        v8h alo = *(const v8h*)(wrow + kb);
        v8h ahi = *(const v8h*)(wrow + kb + 16);
        v16h a = __builtin_shufflevector(alo, ahi, 0,1,2,3,4,5,6,7,
                                                   8,9,10,11,12,13,14,15);
        // B fragment from LDS: N = mrow, K same table order as A
        v16h bf;
        #pragma unroll
        for (int s = 0; s < 16; ++s)
            bf[s] = btile[(kb + (s < 8 ? s : 8 + s)) * 24 + mrow];
        acc = wmma16(a, bf, acc);
    }

    // D layout: VGPR r -> ch = m0 + 8*hi + r ; lane -> pixel = p0 + mrow
    const int chb = m0 + 8 * hi;
    float4 s0 = *(const float4*)(sit + chb);
    float4 s1 = *(const float4*)(sit + chb + 4);
    float4 h0 = *(const float4*)(hit + chb);
    float4 h1 = *(const float4*)(hit + chb + 4);
    v8h oh;
    oh[0] = (_Float16)fmaxf(acc[0] * s0.x + h0.x, 0.f);
    oh[1] = (_Float16)fmaxf(acc[1] * s0.y + h0.y, 0.f);
    oh[2] = (_Float16)fmaxf(acc[2] * s0.z + h0.z, 0.f);
    oh[3] = (_Float16)fmaxf(acc[3] * s0.w + h0.w, 0.f);
    oh[4] = (_Float16)fmaxf(acc[4] * s1.x + h1.x, 0.f);
    oh[5] = (_Float16)fmaxf(acc[5] * s1.y + h1.y, 0.f);
    oh[6] = (_Float16)fmaxf(acc[6] * s1.z + h1.z, 0.f);
    oh[7] = (_Float16)fmaxf(acc[7] * s1.w + h1.w, 0.f);
    const int p = p0 + mrow;
    *(v8h*)(it + ((size_t)b * HW + p) * MIDC + chb) = oh;
}

// ===========================================================================
// Kernel 2: per-wave fused pipeline over 16 voxels.
//   Async-load vf tile -> LDS while doing projection/boxes/bilinear, then
//   WMMA GEMM1 -> LDS fused tile -> WMMA GEMM2 -> BN/ReLU -> out.
//   Block = 128 threads (4 waves), LDS ~53 KB.
// ===========================================================================
__global__ __launch_bounds__(128) void fuse_voxel_kernel(
    const float* __restrict__ vf,          // [N][C3D]
    const int*   __restrict__ coords,      // [N][4]  (b, z, y, x)
    const float* __restrict__ gt_boxes,    // [B][M][7]
    const int*   __restrict__ gt_classes,  // [B][M]
    const _Float16* __restrict__ it,       // [B][HW][MIDC]
    const _Float16* __restrict__ wvtT,     // [C3D][MIDC]
    const _Float16* __restrict__ wfT,      // [2*MIDC][OUTC]
    const float* __restrict__ svt, const float* __restrict__ hvt,
    const float* __restrict__ sf,  const float* __restrict__ hf,
    float* __restrict__ out)               // [N][OUTC]
{
    __shared__ __align__(16) _Float16 fusedAll[4][16 * FSTR];
    __shared__ __align__(16) float    vfTileAll[4][16 * VSTR];
    __shared__ float w3dAll[4][16];
    __shared__ int   sbi[4][16], sx0[4][16], sy0[4][16];
    __shared__ float swx[4][16], swy[4][16];

    const int wave = threadIdx.x >> 5;
    const int lane = threadIdx.x & 31;
    const int n0   = (blockIdx.x * 4 + wave) * 16;
    _Float16* fused = fusedAll[wave];
    float*    w3dv  = w3dAll[wave];
    const int hi   = lane >> 4;
    const int mrow = lane & 15;

    // ---- Stage -1: async DMA of the 16x64-f32 voxel-feature tile to LDS ---
    // Rows padded to VSTR floats so A-fragment ds_load_b128s are bank-clean.
    {
        const uint64_t gbase = (uint64_t)(uintptr_t)(vf + (size_t)n0 * C3D);
        const uint32_t lbase = (uint32_t)(uintptr_t)(&vfTileAll[wave][0]);
        #pragma unroll
        for (int j = 0; j < 8; ++j) {
            const int row = j * 2 + hi;            // 2 rows per issue
            const uint32_t loff = lbase + row * (VSTR * 4) + mrow * 16;
            const uint64_t ga   = gbase + row * (C3D * 4) + mrow * 16;
            asm volatile("global_load_async_to_lds_b128 %0, %1, off"
                         :: "v"(loff), "v"(ga) : "memory");
        }
    }

    // ---- Stage 0: per-voxel projection + class-aware weight (lanes 0-15) --
    if (lane < 16) {
        const int n = n0 + lane;
        const int b = coords[n * 4 + 0];
        const float vz = (float)coords[n * 4 + 1];
        const float vy = (float)coords[n * 4 + 2];
        const float vx = (float)coords[n * 4 + 3];
        const float crx = vx * 0.05f;
        const float cry = vy * 0.05f - 40.0f;
        const float crz = vz * 0.10f - 3.0f;
        const float cx = crx + 0.025f, cy = cry + 0.025f;
        const float px = cx * 10.f + (float)WW * 0.5f;
        const float py = cy * 10.f + (float)HH * 0.5f;
        const float nx = fminf(fmaxf(px / (float)WW * 2.f - 1.f, -1.f), 1.f);
        const float ny = fminf(fmaxf(py / (float)HH * 2.f - 1.f, -1.f), 1.f);
        const float fx = ((nx + 1.f) * (float)WW - 1.f) * 0.5f;
        const float fy = ((ny + 1.f) * (float)HH - 1.f) * 0.5f;
        const float x0f = floorf(fx), y0f = floorf(fy);

        bool any = false; int last = 0;
        const float* gb = gt_boxes + (size_t)b * MBOX * 7;
        for (int m = 0; m < MBOX; ++m) {
            const float* g = gb + m * 7;
            const bool in = (fabsf(crx - g[0]) < g[3] * 0.5f) &
                            (fabsf(cry - g[1]) < g[4] * 0.5f) &
                            (fabsf(crz - g[2]) < g[5] * 0.5f) &
                            (g[3] > 0.f);
            if (in) { any = true; last = m; }
        }
        float w3 = 0.8f;
        if (any) {
            const int c = gt_classes[b * MBOX + last];
            w3 = (c == 0) ? 0.85f : ((c == 1) ? 0.95f : 0.6f);
        }
        w3dv[lane] = w3;
        sbi[wave][lane] = b;
        sx0[wave][lane] = (int)x0f; sy0[wave][lane] = (int)y0f;
        swx[wave][lane] = fx - x0f; swy[wave][lane] = fy - y0f;
    }
    __builtin_amdgcn_wave_barrier();

    // ---- Stage 1: bilinear gather of image features, x (1 - w3d) ----------
    {
        const int ch = lane * 4;             // 32 lanes x 4 ch = 128
        for (int v = 0; v < 16; ++v) {
            const int   b  = sbi[wave][v];
            const int   x0 = sx0[wave][v], y0 = sy0[wave][v];
            const float wx = swx[wave][v], wy = swy[wave][v];
            float a0 = 0.f, a1 = 0.f, a2 = 0.f, a3 = 0.f;
            #pragma unroll
            for (int cnr = 0; cnr < 4; ++cnr) {
                const int xi = x0 + (cnr & 1), yi = y0 + (cnr >> 1);
                float w = ((cnr & 1) ? wx : 1.f - wx) *
                          ((cnr >> 1) ? wy : 1.f - wy);
                if (xi < 0 || xi >= WW || yi < 0 || yi >= HH) w = 0.f;
                const int xc = min(max(xi, 0), WW - 1);
                const int yc = min(max(yi, 0), HH - 1);
                const _Float16* p =
                    it + ((size_t)b * HW + (size_t)yc * WW + xc) * MIDC + ch;
                h4 hv = *(const h4*)p;
                a0 += w * (float)hv[0]; a1 += w * (float)hv[1];
                a2 += w * (float)hv[2]; a3 += w * (float)hv[3];
            }
            const float s = 1.f - w3dv[v];
            h4 st; st[0] = (_Float16)(a0 * s); st[1] = (_Float16)(a1 * s);
            st[2] = (_Float16)(a2 * s);        st[3] = (_Float16)(a3 * s);
            *(h4*)(fused + v * FSTR + MIDC + ch) = st;
        }
    }

    // ---- Stage 2: GEMM1 vt = relu(bn(vf @ w_vt.T)) * w3d -> LDS -----------
    asm volatile("s_wait_asynccnt 0x0" ::: "memory");   // vf tile resident
    {
        const float* vrow = &vfTileAll[wave][mrow * VSTR];
        v16h afr[2];
        #pragma unroll
        for (int kk = 0; kk < 2; ++kk) {
            const int kb = kk * 32 + hi * 8;
            float4 r0 = *(const float4*)(vrow + kb);
            float4 r1 = *(const float4*)(vrow + kb + 4);
            float4 r2 = *(const float4*)(vrow + kb + 16);
            float4 r3 = *(const float4*)(vrow + kb + 20);
            v16h a;
            a[0]=(_Float16)r0.x; a[1]=(_Float16)r0.y; a[2]=(_Float16)r0.z; a[3]=(_Float16)r0.w;
            a[4]=(_Float16)r1.x; a[5]=(_Float16)r1.y; a[6]=(_Float16)r1.z; a[7]=(_Float16)r1.w;
            a[8]=(_Float16)r2.x; a[9]=(_Float16)r2.y; a[10]=(_Float16)r2.z; a[11]=(_Float16)r2.w;
            a[12]=(_Float16)r3.x; a[13]=(_Float16)r3.y; a[14]=(_Float16)r3.z; a[15]=(_Float16)r3.w;
            afr[kk] = a;
        }
        for (int o0 = 0; o0 < MIDC; o0 += 16) {
            v8f acc = {};
            #pragma unroll
            for (int kk = 0; kk < 2; ++kk) {
                const int k = kk * 32 + hi * 8;  // B: lane -> N, K table order
                v16h bf;
                #pragma unroll
                for (int s = 0; s < 16; ++s)
                    bf[s] = wvtT[(size_t)(k + (s < 8 ? s : 8 + s)) * MIDC + o0 + mrow];
                acc = wmma16(afr[kk], bf, acc);
            }
            const int chn = o0 + mrow;           // D: lane -> N (mid-channel)
            const float sc = svt[chn], sh = hvt[chn];
            #pragma unroll
            for (int r = 0; r < 8; ++r) {
                const int vox = r + 8 * hi;      // D: VGPR -> M (voxel)
                const float y = fmaxf(acc[r] * sc + sh, 0.f) * w3dv[vox];
                fused[vox * FSTR + chn] = (_Float16)y;
            }
        }
    }
    __builtin_amdgcn_wave_barrier();

    // ---- Stage 3: GEMM2 out = relu(bn(fused @ w_f.T)) ----------------------
    {
        const _Float16* ap = fused + mrow * FSTR;
        for (int o0 = 0; o0 < OUTC; o0 += 16) {
            v8f acc = {};
            for (int k0 = 0; k0 < 2 * MIDC; k0 += 32) {
                const int kb = k0 + hi * 8;
                v8h lo = *(const v8h*)(ap + kb);
                v8h hx = *(const v8h*)(ap + kb + 16);
                v16h a = __builtin_shufflevector(lo, hx, 0,1,2,3,4,5,6,7,
                                                         8,9,10,11,12,13,14,15);
                v16h bf;
                #pragma unroll
                for (int s = 0; s < 16; ++s)
                    bf[s] = wfT[(size_t)(kb + (s < 8 ? s : 8 + s)) * OUTC + o0 + mrow];
                acc = wmma16(a, bf, acc);
            }
            const int chn = o0 + mrow;
            const float sc = sf[chn], sh = hf[chn];
            #pragma unroll
            for (int r = 0; r < 8; ++r) {
                const int vox = r + 8 * hi;
                out[(size_t)(n0 + vox) * OUTC + chn] =
                    fmaxf(acc[r] * sc + sh, 0.f);
            }
        }
    }
}

// ===========================================================================
extern "C" void kernel_launch(void* const* d_in, const int* in_sizes, int n_in,
                              void* d_out, int out_size, void* d_ws, size_t ws_size,
                              hipStream_t stream)
{
    const float* vf      = (const float*)d_in[0];
    const int*   coords  = (const int*)d_in[1];
    const float* img     = (const float*)d_in[2];
    const float* gtb     = (const float*)d_in[3];
    const int*   gtc     = (const int*)d_in[4];
    const float* w_vt    = (const float*)d_in[5];
    const float* b_vt    = (const float*)d_in[6];
    const float* g_vt    = (const float*)d_in[7];
    const float* be_vt   = (const float*)d_in[8];
    const float* m_vt    = (const float*)d_in[9];
    const float* v_vt    = (const float*)d_in[10];
    const float* w_it    = (const float*)d_in[11];
    const float* b_it    = (const float*)d_in[12];
    const float* g_it    = (const float*)d_in[13];
    const float* be_it   = (const float*)d_in[14];
    const float* m_it    = (const float*)d_in[15];
    const float* v_it    = (const float*)d_in[16];
    const float* w_f     = (const float*)d_in[17];
    const float* b_f     = (const float*)d_in[18];
    const float* g_f     = (const float*)d_in[19];
    const float* be_f    = (const float*)d_in[20];
    const float* m_f     = (const float*)d_in[21];
    const float* v_f     = (const float*)d_in[22];
    float* out = (float*)d_out;

    // Workspace layout (bytes), all offsets 256-aligned.
    char* ws = (char*)d_ws;
    size_t off = 0;
    _Float16* it_h   = (_Float16*)(ws + off); off += (size_t)BB * HW * MIDC * 2;   // 30.7 MB
    _Float16* w_it_h = (_Float16*)(ws + off); off += (size_t)MIDC * C2D * 2;
    _Float16* wvtT   = (_Float16*)(ws + off); off += (size_t)C3D * MIDC * 2;
    _Float16* wfT    = (_Float16*)(ws + off); off += (size_t)2 * MIDC * OUTC * 2;
    float* svt = (float*)(ws + off); off += 512;
    float* hvt = (float*)(ws + off); off += 512;
    float* sit = (float*)(ws + off); off += 512;
    float* hit = (float*)(ws + off); off += 512;
    float* sfv = (float*)(ws + off); off += 512;
    float* hfv = (float*)(ws + off); off += 512;
    (void)in_sizes; (void)n_in; (void)out_size; (void)ws_size;

    prep_kernel<<<96, 256, 0, stream>>>(
        w_vt, w_it, w_f,
        b_vt, g_vt, be_vt, m_vt, v_vt,
        b_it, g_it, be_it, m_it, v_it,
        b_f, g_f, be_f, m_f, v_f,
        w_it_h, wvtT, wfT, svt, hvt, sit, hit, sfv, hfv);

    img_xform_kernel<<<BB * (HW / 16), 256, 0, stream>>>(
        img, w_it_h, sit, hit, it_h);

    fuse_voxel_kernel<<<NVOX / 64, 128, 0, stream>>>(
        vf, coords, gtb, gtc, it_h, wvtT, wfT, svt, hvt, sfv, hfv, out);
}
